// LIFLayer_29454885716187
// MI455X (gfx1250) — compile-verified, matrix operand order checked by reading, not run on hardware
//
#include <hip/hip_runtime.h>
#include <hip/hip_bf16.h>

#define N_NEUR 4096
#define T_STEPS 300
#define T_PAD 304
#define THRESH_V 10.0f
#define DECAY_V 0.95f
#define REFRAC_V 1.0f
#define DT_V 1.0f
#define TIME_V 300.0f

typedef __attribute__((ext_vector_type(16))) __bf16          v16bf;
typedef __attribute__((ext_vector_type(16))) unsigned short  v16us;
typedef __attribute__((ext_vector_type(8)))  unsigned short  v8us;
typedef __attribute__((ext_vector_type(8)))  float           v8f;

// float -> bf16 bits, round-to-nearest-even
__device__ __forceinline__ unsigned short f2bf_bits(float f) {
  union { float f; unsigned u; } x;
  x.f = f;
  unsigned u = x.u + 0x7FFFu + ((x.u >> 16) & 1u);
  return (unsigned short)(u >> 16);
}

// ---------------------------------------------------------------------------
// Prep 1: pad+convert inp -> a_bf[304][4096] (bf16 bits; rows >= 300 are zero)
// ---------------------------------------------------------------------------
__global__ void prep_a_kernel(const float* __restrict__ inp,
                              unsigned short* __restrict__ a_bf) {
  const size_t idx = (size_t)blockIdx.x * blockDim.x + threadIdx.x;
  if (idx >= (size_t)T_PAD * N_NEUR) return;
  const size_t m = idx / N_NEUR;
  a_bf[idx] = (m < T_STEPS) ? f2bf_bits(inp[idx]) : (unsigned short)0;
}

// ---------------------------------------------------------------------------
// Prep 2: transpose+convert w[k][n] -> wt_bf[n][k] (bf16 bits), 32x32 LDS tiles
// ---------------------------------------------------------------------------
__global__ void __launch_bounds__(256) prep_wt_kernel(const float* __restrict__ w,
                                                      unsigned short* __restrict__ wt_bf) {
  __shared__ unsigned short tile[32][33];          // +1 pad: no bank conflicts
  const int tx = threadIdx.x & 31;                 // column within tile
  const int ty4 = threadIdx.x >> 5;                // 0..7, each covers 4 rows
  const int k0 = (blockIdx.x & 127) * 32;          // 4096/32 = 128 tiles per dim
  const int n0 = (blockIdx.x >> 7) * 32;
#pragma unroll
  for (int r = 0; r < 4; ++r) {
    const int k = k0 + ty4 * 4 + r;
    tile[ty4 * 4 + r][tx] = f2bf_bits(w[(size_t)k * N_NEUR + n0 + tx]);
  }
  __syncthreads();
#pragma unroll
  for (int r = 0; r < 4; ++r) {
    const int n = n0 + ty4 * 4 + r;
    wt_bf[(size_t)n * N_NEUR + k0 + tx] = tile[tx][ty4 * 4 + r];
  }
}

// ---------------------------------------------------------------------------
// Feedforward GEMM: FF = inp @ w, 300x4096x4096, bf16 in / f32 accumulate.
// One wave32 per 16x16 tile; K unrolled by 64 into two WMMA chains.
// A and wT are row-contiguous in K -> fragments are two b128 loads each.
// ---------------------------------------------------------------------------
__global__ void __launch_bounds__(256) ff_gemm_kernel(const unsigned short* __restrict__ a_bf,
                                                      const unsigned short* __restrict__ wt_bf,
                                                      float* __restrict__ ff) {
  const int lane    = threadIdx.x & 31;
  const int wave_id = blockIdx.x * 8 + (threadIdx.x >> 5);
  const int mtile   = wave_id >> 8;                // / 256 N-tiles
  const int ntile   = wave_id & 255;

  const int row16 = lane & 15;
  const int khalf = (lane >> 4) << 3;              // lanes 0-15: K+0, 16-31: K+8
  const unsigned short* ap = a_bf  + (size_t)(mtile * 16 + row16) * N_NEUR + khalf;
  const unsigned short* bp = wt_bf + (size_t)(ntile * 16 + row16) * N_NEUR + khalf;

  v8f acc0 = {};
  v8f acc1 = {};
  for (int k0 = 0; k0 < N_NEUR; k0 += 64) {
    // fragment = {K+0..7, K+16..23} relative to (k0 + khalf)
    v8us a0l = *(const v8us*)(ap + k0);
    v8us a0h = *(const v8us*)(ap + k0 + 16);
    v8us b0l = *(const v8us*)(bp + k0);
    v8us b0h = *(const v8us*)(bp + k0 + 16);
    v8us a1l = *(const v8us*)(ap + k0 + 32);
    v8us a1h = *(const v8us*)(ap + k0 + 48);
    v8us b1l = *(const v8us*)(bp + k0 + 32);
    v8us b1h = *(const v8us*)(bp + k0 + 48);
    v16us au0 = __builtin_shufflevector(a0l, a0h, 0,1,2,3,4,5,6,7,8,9,10,11,12,13,14,15);
    v16us bu0 = __builtin_shufflevector(b0l, b0h, 0,1,2,3,4,5,6,7,8,9,10,11,12,13,14,15);
    v16us au1 = __builtin_shufflevector(a1l, a1h, 0,1,2,3,4,5,6,7,8,9,10,11,12,13,14,15);
    v16us bu1 = __builtin_shufflevector(b1l, b1h, 0,1,2,3,4,5,6,7,8,9,10,11,12,13,14,15);
    acc0 = __builtin_amdgcn_wmma_f32_16x16x32_bf16(
        false, __builtin_bit_cast(v16bf, au0), false, __builtin_bit_cast(v16bf, bu0),
        (short)0, acc0, false, false);
    acc1 = __builtin_amdgcn_wmma_f32_16x16x32_bf16(
        false, __builtin_bit_cast(v16bf, au1), false, __builtin_bit_cast(v16bf, bu1),
        (short)0, acc1, false, false);
  }

  const int bcol = ntile * 16 + row16;
#pragma unroll
  for (int r = 0; r < 8; ++r) {
    // f32 C/D layout: VGPR r, lanes 0-15 -> M=r, lanes 16-31 -> M=r+8
    const int m = mtile * 16 + r + ((lane >> 4) << 3);
    if (m < T_STEPS) ff[(size_t)m * N_NEUR + bcol] = acc0[r] + acc1[r];
  }
}

// ---------------------------------------------------------------------------
// Init: zero all (T+1) spike bitmasks, membrane v, refractory rt (every call).
// ---------------------------------------------------------------------------
__global__ void init_kernel(unsigned* __restrict__ masks,
                            float* __restrict__ v,
                            float* __restrict__ rt) {
  const int idx = blockIdx.x * blockDim.x + threadIdx.x;
  const int total_mask = (T_STEPS + 1) * 128;
  if (idx < total_mask) masks[idx] = 0u;
  if (idx < N_NEUR) { v[idx] = 0.0f; rt[idx] = 0.0f; }
}

// ---------------------------------------------------------------------------
// One LIF timestep. Deterministic event-driven recurrence:
//   masks[t] -> ascending LDS index list (popcount + exclusive scan),
//   8 partial lanes per neuron with fixed stride-8 partition + fixed-order
//   reduction, spikes OR'd into masks[t+1].
// ---------------------------------------------------------------------------
__global__ void __launch_bounds__(512) lif_step_kernel(int t,
                                                       const float* __restrict__ w_rec,
                                                       const float* __restrict__ ff,
                                                       unsigned* __restrict__ masks,
                                                       float* __restrict__ v,
                                                       float* __restrict__ rt,
                                                       int* __restrict__ s_out,
                                                       float* __restrict__ t_out,
                                                       float* __restrict__ v_fin) {
  __shared__ unsigned smask[128];
  __shared__ int      woff[129];
  __shared__ int      idxs[N_NEUR];
  __shared__ float    partial[8][64];

  const int tid = threadIdx.x;
  const int jn  = tid & 63;                        // neuron within block
  const int sub = tid >> 6;                        // partial lane 0..7
  const int j   = blockIdx.x * 64 + jn;

  const unsigned* cur = masks + (size_t)t * 128;
  unsigned*       nxt = masks + (size_t)(t + 1) * 128;

  if (tid < 128) smask[tid] = cur[tid];
  __syncthreads();
  if (tid == 0) {                                  // serial exclusive scan (128 words)
    int acc = 0;
#pragma unroll 4
    for (int i = 0; i < 128; ++i) { woff[i] = acc; acc += __popc(smask[i]); }
    woff[128] = acc;
  }
  __syncthreads();
  if (tid < 128) {                                 // expand bits -> ascending index list
    unsigned b = smask[tid];
    int o = woff[tid];
    while (b) {
      const int kbit = __ffs(b) - 1;
      b &= b - 1u;
      idxs[o++] = (tid << 5) + kbit;
    }
  }
  __syncthreads();

  const int cnt = woff[128];
  float xp = 0.0f;
#pragma unroll 4
  for (int p = sub; p < cnt; p += 8)               // fixed partition, ascending order
    xp += w_rec[(size_t)idxs[p] * N_NEUR + j];
  partial[sub][jn] = xp;
  __syncthreads();

  if (sub == 0) {
    float x = ff[(size_t)t * N_NEUR + j];
#pragma unroll
    for (int s = 0; s < 8; ++s) x += partial[s][jn];

    // LIF update, mirroring reference step():
    float vj = DECAY_V * v[j];
    float rj = rt[j];
    if (rj > 0.0f) x = 0.0f;
    rj -= DT_V;
    vj += x;
    const bool spike = (vj >= THRESH_V);
    const float tv = spike ? ((float)t * DT_V / TIME_V) : -__builtin_inff();
    if (spike) { rj = REFRAC_V; vj = 0.0f; }

    v[j]  = vj;
    rt[j] = rj;
    s_out[(size_t)t * N_NEUR + j] = spike ? 1 : 0;
    t_out[(size_t)t * N_NEUR + j] = tv;
    if (spike) atomicOr(&nxt[j >> 5], 1u << (j & 31));
    if (t == T_STEPS - 1) v_fin[j] = vj;
  }
}

// ---------------------------------------------------------------------------
extern "C" void kernel_launch(void* const* d_in, const int* in_sizes, int n_in,
                              void* d_out, int out_size, void* d_ws, size_t ws_size,
                              hipStream_t stream) {
  (void)in_sizes; (void)n_in; (void)out_size; (void)ws_size;

  const float* inp   = (const float*)d_in[0];      // [T, N]
  const float* w     = (const float*)d_in[1];      // [N, N]
  const float* w_rec = (const float*)d_in[2];      // [N, N]

  // Workspace layout (~41 MB)
  float*          ff    = (float*)d_ws;                                   // T*N f32
  unsigned short* a_bf  = (unsigned short*)(ff + (size_t)T_STEPS * N_NEUR);     // 304*4096 bf16
  unsigned short* wt_bf = a_bf + (size_t)T_PAD * N_NEUR;                  // 4096*4096 bf16
  unsigned*       masks = (unsigned*)(wt_bf + (size_t)N_NEUR * N_NEUR);   // (T+1)*128 u32
  float*          v     = (float*)(masks + (size_t)(T_STEPS + 1) * 128);  // N f32
  float*          rtbuf = v + N_NEUR;                                     // N f32

  // Outputs: s_all (int32 T*N) ++ t_all (f32 T*N) ++ v_fin (f32 N)
  int*   s_out = (int*)d_out;
  float* t_out = (float*)d_out + (size_t)T_STEPS * N_NEUR;
  float* v_fin = (float*)d_out + (size_t)2 * T_STEPS * N_NEUR;

  const int mask_words = (T_STEPS + 1) * 128;
  init_kernel<<<(mask_words + 255) / 256, 256, 0, stream>>>(masks, v, rtbuf);

  const size_t a_elems = (size_t)T_PAD * N_NEUR;
  prep_a_kernel<<<(int)((a_elems + 255) / 256), 256, 0, stream>>>(inp, a_bf);
  prep_wt_kernel<<<128 * 128, 256, 0, stream>>>(w, wt_bf);

  // 19 M-tiles x 256 N-tiles = 4864 waves; 8 waves/block -> 608 blocks
  const int waves_total = ((T_STEPS + 15) / 16) * (N_NEUR / 16);
  ff_gemm_kernel<<<waves_total / 8, 256, 0, stream>>>(a_bf, wt_bf, ff);

  // Sequential recurrence: one stream-ordered launch per timestep.
  for (int t = 0; t < T_STEPS; ++t) {
    lif_step_kernel<<<N_NEUR / 64, 512, 0, stream>>>(
        t, w_rec, ff, masks, v, rtbuf, s_out, t_out, v_fin);
  }
}
